// HyperNet_76158360093293
// MI455X (gfx1250) — matrix operand order repaired and checked by reading, not compiled, not run on hardware
//
#include <hip/hip_runtime.h>
#include <hip/hip_bf16.h>

#define NSAMP   10
#define IN_DIM  512
#define HID_DIM 1024
#define OUT_DIM 512
#define BATCH   1024
#define N1      (HID_DIM * IN_DIM)              // 524288
#define N2      (OUT_DIM * HID_DIM)             // 524288
#define NPARAMS (N1 + N2)                       // 1048576

typedef __bf16 bf16_t;
typedef __attribute__((ext_vector_type(16))) __bf16 v16bf;
typedef __attribute__((ext_vector_type(8)))  __bf16 v8bf;
typedef __attribute__((ext_vector_type(8)))  float  v8f;

__device__ __forceinline__ float elu_f(float x) {
    return x > 0.0f ? x : (__expf(x) - 1.0f);
}

// ---------------- Stage 0: tiny hypernet MLP  h = elu(elu(z Wa^T) Wb^T) ----
__global__ __launch_bounds__(128)
void hyper_mlp(const float* __restrict__ z, const float* __restrict__ Wa,
               const float* __restrict__ Wb, float* __restrict__ h_out) {
    __shared__ float zs[80], was[80], wbs[100], h1[100];
    int t = threadIdx.x;
    if (t < 80)  { zs[t] = z[t]; was[t] = Wa[t]; }
    if (t < 100) { wbs[t] = Wb[t]; }
    __syncthreads();
    if (t < 100) {
        int s = t / 10, w = t % 10;
        float acc = 0.0f;
        #pragma unroll
        for (int c = 0; c < 8; ++c) acc += zs[s * 8 + c] * was[w * 8 + c];
        h1[t] = elu_f(acc);
    }
    __syncthreads();
    if (t < 100) {
        int s = t / 10, v = t % 10;
        float acc = 0.0f;
        #pragma unroll
        for (int w = 0; w < 10; ++w) acc += h1[s * 10 + w] * wbs[v * 10 + w];
        h_out[t] = elu_f(acc);
    }
}

// ---------------- Stage 1a: x f32 -> bf16 ---------------------------------
__global__ __launch_bounds__(256)
void cvt_x(const float* __restrict__ x, bf16_t* __restrict__ xb, int npairs) {
    int i = blockIdx.x * blockDim.x + threadIdx.x;
    if (i < npairs) {
        float2 v = ((const float2*)x)[i];
        xb[2 * i]     = (bf16_t)v.x;
        xb[2 * i + 1] = (bf16_t)v.y;
    }
}

// ---------------- Stage 1b: theta = h Wc^T, emitted as bf16 W1/W2 ----------
// One thread handles 2 consecutive params p; Wc row read once, reused for
// all 10 samples (Wc is the dominant HBM stream: 40 MB).
__global__ __launch_bounds__(256)
void theta_gen(const float* __restrict__ Wc, const float* __restrict__ h,
               bf16_t* __restrict__ w1b, bf16_t* __restrict__ w2b) {
    __shared__ float hs[NSAMP * 10];
    if (threadIdx.x < NSAMP * 10) hs[threadIdx.x] = h[threadIdx.x];
    __syncthreads();
    long p0 = ((long)blockIdx.x * blockDim.x + threadIdx.x) * 2;
    if (p0 >= NPARAMS) return;
    float wc0[10], wc1[10];
    const float* r = Wc + p0 * 10;
    #pragma unroll
    for (int v = 0; v < 10; ++v) { wc0[v] = r[v]; wc1[v] = r[10 + v]; }
    bool inW1     = (p0 < N1);
    long idx      = inW1 ? p0 : (p0 - N1);
    long strideS  = inW1 ? (long)N1 : (long)N2;
    bf16_t* base  = inW1 ? w1b : w2b;
    #pragma unroll
    for (int s = 0; s < NSAMP; ++s) {
        float t0 = 0.0f, t1 = 0.0f;
        #pragma unroll
        for (int v = 0; v < 10; ++v) {
            t0 += hs[s * 10 + v] * wc0[v];
            t1 += hs[s * 10 + v] * wc1[v];
        }
        bf16_t* d = base + s * strideS + idx;
        d[0] = (bf16_t)t0;
        d[1] = (bf16_t)t1;
    }
}

// ---------------- WMMA fragment loads (ISA 16-bit layouts, wave32) ---------
// A 16x32: lanes 0-15 row M=lane hold K=k..k+7 (v0..3) and K=k+16..k+23
// (v4..7); lanes 16-31 hold the +8 K-halves.
__device__ __forceinline__ v16bf load_frag_a(const bf16_t* __restrict__ tile,
                                             int ld, int k, int lane) {
    int m = lane & 15, h = lane >> 4;
    const bf16_t* p = tile + (long)m * ld + k + h * 8;
    v8bf lo = *(const v8bf*)p;          // K = k + h*8 .. +7
    v8bf hi = *(const v8bf*)(p + 16);   // K = k + 16 + h*8 .. +7
    v16bf f;
    #pragma unroll
    for (int i = 0; i < 8; ++i) { f[i] = lo[i]; f[i + 8] = hi[i]; }
    return f;
}
// B 32x16: lane = column N (0-15); lanes 0-15 hold K=k..k+7 and K=k+16..k+23,
// lanes 16-31 hold the +8 K-halves (same order as A).
__device__ __forceinline__ v16bf load_frag_b(const bf16_t* __restrict__ tile,
                                             int ld, int k, int lane) {
    int n = lane & 15, h = lane >> 4;
    const bf16_t* p = tile + (long)n * ld + k + h * 8;
    v8bf lo = *(const v8bf*)p;
    v8bf hi = *(const v8bf*)(p + 16);
    v16bf f;
    #pragma unroll
    for (int i = 0; i < 8; ++i) { f[i] = lo[i]; f[i + 8] = hi[i]; }
    return f;
}

// ---------------- Stages 2/3: NT GEMM, 32x32 per wave, 64x128 per block ----
// ELU_BF16OUT=true : D = bf16(elu(A B^T))   (layer-1 activations)
// ELU_BF16OUT=false: D = f32(A B^T)         (final output)
template <bool ELU_BF16OUT>
__global__ __launch_bounds__(256)
void gemm_nt(const bf16_t* __restrict__ A, const bf16_t* __restrict__ B,
             void* __restrict__ Out, int N, int K,
             long aStrideS, long bStrideS, long oStrideS) {
    int s    = blockIdx.z;
    int lane = threadIdx.x & 31;
    int wid  = threadIdx.x >> 5;          // 8 waves
    int mb   = blockIdx.y * 64  + (wid >> 2) * 32;
    int nb   = blockIdx.x * 128 + (wid & 3) * 32;

    const bf16_t* a  = A + (long)s * aStrideS;
    const bf16_t* b  = B + (long)s * bStrideS;
    const bf16_t* a0 = a + (long)mb * K;
    const bf16_t* a1 = a + (long)(mb + 16) * K;
    const bf16_t* b0 = b + (long)nb * K;
    const bf16_t* b1 = b + (long)(nb + 16) * K;

    v8f c00 = {}, c01 = {}, c10 = {}, c11 = {};
    for (int k = 0; k < K; k += 32) {
        v16bf fa0 = load_frag_a(a0, K, k, lane);
        v16bf fa1 = load_frag_a(a1, K, k, lane);
        v16bf fb0 = load_frag_b(b0, K, k, lane);
        v16bf fb1 = load_frag_b(b1, K, k, lane);
        c00 = __builtin_amdgcn_wmma_f32_16x16x32_bf16(false, fa0, false, fb0, (short)0, c00, false, false);
        c01 = __builtin_amdgcn_wmma_f32_16x16x32_bf16(false, fa0, false, fb1, (short)0, c01, false, false);
        c10 = __builtin_amdgcn_wmma_f32_16x16x32_bf16(false, fa1, false, fb0, (short)0, c10, false, false);
        c11 = __builtin_amdgcn_wmma_f32_16x16x32_bf16(false, fa1, false, fb1, (short)0, c11, false, false);
    }

    // C/D layout: VGPR r -> row = r + 8*(lane>>4), col = lane&15.
    int half = lane >> 4, nl = lane & 15;
    if constexpr (ELU_BF16OUT) {
        bf16_t* o = (bf16_t*)Out + (long)s * oStrideS;
        #pragma unroll
        for (int r = 0; r < 8; ++r) {
            int rr0 = mb + r + 8 * half, rr1 = rr0 + 16;
            o[(long)rr0 * N + nb + nl]      = (bf16_t)elu_f(c00[r]);
            o[(long)rr0 * N + nb + 16 + nl] = (bf16_t)elu_f(c01[r]);
            o[(long)rr1 * N + nb + nl]      = (bf16_t)elu_f(c10[r]);
            o[(long)rr1 * N + nb + 16 + nl] = (bf16_t)elu_f(c11[r]);
        }
    } else {
        float* o = (float*)Out + (long)s * oStrideS;
        #pragma unroll
        for (int r = 0; r < 8; ++r) {
            int rr0 = mb + r + 8 * half, rr1 = rr0 + 16;
            o[(long)rr0 * N + nb + nl]      = c00[r];
            o[(long)rr0 * N + nb + 16 + nl] = c01[r];
            o[(long)rr1 * N + nb + nl]      = c10[r];
            o[(long)rr1 * N + nb + 16 + nl] = c11[r];
        }
    }
}

// ---------------- Host-side orchestration ----------------------------------
extern "C" void kernel_launch(void* const* d_in, const int* in_sizes, int n_in,
                              void* d_out, int out_size, void* d_ws, size_t ws_size,
                              hipStream_t stream) {
    const float* x  = (const float*)d_in[0];
    const float* z  = (const float*)d_in[1];
    const float* Wa = (const float*)d_in[2];
    const float* Wb = (const float*)d_in[3];
    const float* Wc = (const float*)d_in[4];
    float* y = (float*)d_out;

    // Workspace layout (all 16B-aligned):
    //   h    : 100 f32                     @ 0        (pad to 1 KB)
    //   xb   : 1024*512 bf16 (1 MB)        @ 1024
    //   w1b  : 10*1024*512 bf16 (10 MB)    @ 1 KB + 1 MB
    //   w2b  : 10*512*1024 bf16 (10 MB)    @ + w1b
    //   ab   : 10*1024*1024 bf16 (20 MB)   @ + w2b     => ~42 MB total
    char* ws = (char*)d_ws;
    float*  h   = (float*)ws;
    bf16_t* xb  = (bf16_t*)(ws + 1024);
    bf16_t* w1b = (bf16_t*)(ws + 1024 + (size_t)BATCH * IN_DIM * 2);
    bf16_t* w2b = (bf16_t*)((char*)w1b + (size_t)NSAMP * N1 * 2);
    bf16_t* ab  = (bf16_t*)((char*)w2b + (size_t)NSAMP * N2 * 2);

    hyper_mlp<<<1, 128, 0, stream>>>(z, Wa, Wb, h);

    int npairs = BATCH * IN_DIM / 2;
    cvt_x<<<(npairs + 255) / 256, 256, 0, stream>>>(x, xb, npairs);

    theta_gen<<<NPARAMS / 2 / 256, 256, 0, stream>>>(Wc, h, w1b, w2b);

    // a[s] = elu(x W1[s]^T): M=1024, N=1024, K=512  (x shared: aStrideS=0)
    dim3 g1(HID_DIM / 128, BATCH / 64, NSAMP);
    gemm_nt<true><<<g1, 256, 0, stream>>>(xb, w1b, (void*)ab, HID_DIM, IN_DIM,
                                          0L, (long)N1, (long)BATCH * HID_DIM);

    // y[s] = a[s] W2[s]^T: M=1024, N=512, K=1024
    dim3 g2(OUT_DIM / 128, BATCH / 64, NSAMP);
    gemm_nt<false><<<g2, 256, 0, stream>>>(ab, w2b, (void*)y, OUT_DIM, HID_DIM,
                                           (long)BATCH * HID_DIM, (long)N2,
                                           (long)BATCH * OUT_DIM);
}
